// PaddedLoraB_59459527246474
// MI455X (gfx1250) — matrix operand order repaired and checked by reading, not compiled, not run on hardware
//
#include <hip/hip_runtime.h>

typedef __attribute__((ext_vector_type(16))) _Float16 v16h;
typedef __attribute__((ext_vector_type(8)))  float    v8f;

#define BATCH   128
#define RLORA   64
#define HIDDEN  4096
#define NSPLIT  8                           // column-slices per sample (grid.y)
#define COLS_PER_BLOCK (HIDDEN / NSPLIT)    // 512
#define COLS_PER_WAVE  (COLS_PER_BLOCK / 8) // 64 halfs = 128B = 1 cacheline/lane-row
#define NTILES (COLS_PER_WAVE / 16)         // 4 N-tiles of 16

// One block = one (sample, column-slice). 256 threads = 8 wave32s.
// Each wave computes 64 output columns via v_wmma_f32_16x16x32_f16,
// A = y in row M=0 (rows 1..15 zero), B = weight tile (lane L = N-column L%16).
__global__ __launch_bounds__(256) void lora_b_wmma_kernel(
    const _Float16* __restrict__ y,      // [BATCH, 1, RLORA]
    const int*      __restrict__ wids,   // [BATCH]
    const _Float16* __restrict__ loraB,  // [NUM_ADAPTERS, RLORA, HIDDEN]
    _Float16*       __restrict__ out)    // [BATCH, 1, HIDDEN]
{
    const int b    = blockIdx.x;
    const int slab = blockIdx.y;
    const int lane = threadIdx.x & 31;
    const int wave = threadIdx.x >> 5;

    const int wid = wids[b];                             // uniform -> s_load
    const _Float16* __restrict__ W  = loraB + (size_t)wid * (RLORA * HIDDEN);
    const _Float16* __restrict__ yb = y + b * RLORA;

    // Per-lane row pointers K-row L / K-row 32+L (unused; the loads below index W directly).
    const _Float16* __restrict__ Wk0 = W + (size_t)lane        * HIDDEN;
    const _Float16* __restrict__ Wk1 = W + (size_t)(32 + lane) * HIDDEN;

    const int n_base = slab * COLS_PER_BLOCK + wave * COLS_PER_WAVE;

    // ---- Stage 1: issue ALL weight-fragment loads (2 chunks x 4 tiles x 16 = 128 16-bit loads per lane).
    // Per lane & K-row the 4 tiles cover one aligned 128B line -> full-line HBM use.
    v16h bf0[NTILES], bf1[NTILES];
    #pragma unroll
    for (int t = 0; t < NTILES; ++t) {
        #pragma unroll
        for (int s = 0; s < 16; ++s) {
            const int kk = 8 * (lane >> 4) + (s < 8 ? s : s + 8);
            bf0[t][s] = W[(size_t)kk * HIDDEN + n_base + t * 16 + (lane & 15)];
            bf1[t][s] = W[(size_t)(32 + kk) * HIDDEN + n_base + t * 16 + (lane & 15)];
        }
    }

    // ---- A fragments for the two K=32 chunks: row M=0 carries y, rest 0.
    // 16-bit A 16x32 layout: lanes 0-15 (M=0-15) hold K=0..7 (elems 0-7) and
    // K=16..23 (elems 8-15); lanes 16-31 hold K=8..15 and K=24..31.
    v16h a0 = {}, a1 = {};
    if (lane == 0 || lane == 16) {
        const int koff = (lane == 16) ? 8 : 0;
        #pragma unroll
        for (int j = 0; j < 8; ++j) {
            a0[j]     = yb[ 0 + koff + j];
            a0[j + 8] = yb[16 + koff + j];
            a1[j]     = yb[32 + koff + j];
            a1[j + 8] = yb[48 + koff + j];
        }
    }

    // ---- Stage 2: 8 WMMAs, independent accumulator chain per tile.
    float r[NTILES];
    #pragma unroll
    for (int t = 0; t < NTILES; ++t) {
        v8f c = {};
        c = __builtin_amdgcn_wmma_f32_16x16x32_f16(
                /*neg_a=*/false, a0, /*neg_b=*/false, bf0[t],
                /*c_mod=*/(short)0, c, /*reuse_a=*/false, /*reuse_b=*/false);
        c = __builtin_amdgcn_wmma_f32_16x16x32_f16(
                false, a1, false, bf1[t], (short)0, c, false, false);
        r[t] = c[0];                         // D row M=0 lives on lanes 0..15
    }

    // ---- Stage 3: single EXEC region, 4 clause-able b16 stores.
    if (lane < 16) {
        _Float16* __restrict__ outb = out + b * HIDDEN + n_base + lane;
        #pragma unroll
        for (int t = 0; t < NTILES; ++t) {
            outb[t * 16] = (_Float16)(2.0f * r[t]);
        }
    }
}

extern "C" void kernel_launch(void* const* d_in, const int* in_sizes, int n_in,
                              void* d_out, int out_size, void* d_ws, size_t ws_size,
                              hipStream_t stream) {
    const _Float16* y     = (const _Float16*)d_in[0];
    const int*      wids  = (const int*)d_in[1];
    const _Float16* loraB = (const _Float16*)d_in[2];
    _Float16*       out   = (_Float16*)d_out;

    dim3 grid(BATCH, NSPLIT, 1);   // 1024 workgroups
    dim3 block(256, 1, 1);         // 8 wave32s
    lora_b_wmma_kernel<<<grid, block, 0, stream>>>(y, wids, loraB, out);
}
